// MultiHeadAttention_40398462386802
// MI455X (gfx1250) — compile-verified, matrix operand order checked by reading, not run on hardware
//
#include <hip/hip_runtime.h>

#define D_MODEL 768
#define SEQ     2048
#define BATCH   4
#define NHEADS  4
#define HDIM    192          // D_MODEL / NHEADS
#define NTOK    (BATCH * SEQ)
#define NEGV    (-1.0e9f)    // reference NEG
#define INV_SCALE (1.0f / 20.0f)
#define LN_EPS  1e-5f

typedef __bf16 bf16_t;
typedef __attribute__((ext_vector_type(16))) __bf16 v16bf;
typedef __attribute__((ext_vector_type(8)))  float  v8f;
typedef __attribute__((ext_vector_type(4)))  float  f32x4;
typedef int i32x4 __attribute__((vector_size(16)));   // matches builtin param

#define AS1 __attribute__((address_space(1)))
#define AS3 __attribute__((address_space(3)))

#if defined(__has_builtin)
#if __has_builtin(__builtin_amdgcn_global_load_async_to_lds_b128)
#define HAVE_ASYNC_LDS 1
#endif
#endif
#ifndef HAVE_ASYNC_LDS
#define HAVE_ASYNC_LDS 0
#endif

// 16-byte global -> LDS copy.  Async (ASYNCcnt-tracked, no VGPR round trip)
// when the gfx1250 builtin is available; plain b128 copy otherwise.
static __device__ __forceinline__ void cp16_g2l(const bf16_t* g, bf16_t* l) {
#if HAVE_ASYNC_LDS
  __builtin_amdgcn_global_load_async_to_lds_b128(
      (AS1 i32x4*)(unsigned long long)g,   // global AS is numerically flat
      (AS3 i32x4*)l, 0, 0);
#else
  *(f32x4*)l = *(const f32x4*)g;
#endif
}

static __device__ __forceinline__ void async_wait0() {
#if HAVE_ASYNC_LDS
#if __has_builtin(__builtin_amdgcn_s_wait_asynccnt)
  __builtin_amdgcn_s_wait_asynccnt(0);
#else
  asm volatile("s_wait_asynccnt 0" ::: "memory");
#endif
#endif
}

static __device__ __forceinline__ bf16_t f2bf(float f) {
  union { float f; unsigned u; } a; a.f = f;
  unsigned r = a.u + 0x7FFFu + ((a.u >> 16) & 1u);   // round-to-nearest-even
  union { unsigned short s; bf16_t b; } o; o.s = (unsigned short)(r >> 16);
  return o.b;
}

static __device__ __forceinline__ v8f vzero8() {
  v8f z;
#pragma unroll
  for (int i = 0; i < 8; ++i) z[i] = 0.0f;
  return z;
}

// Gather a 16x32 bf16 fragment (A-layout; also used for B via its column-major
// symmetry): lane L in [0,15] holds row/col r=L with K = 0..7 and 16..23;
// lanes 16..31 hold K = 8..15 and 24..31.  Two b128 loads per lane.
static __device__ __forceinline__ v16bf load_frag(const bf16_t* base, int ld, int lane) {
  const int r    = lane & 15;
  const int koff = (lane >> 4) * 8;
  const bf16_t* p = base + (size_t)r * ld + koff;
  union { f32x4 f[2]; v16bf v; } u;
  u.f[0] = *(const f32x4*)(p);
  u.f[1] = *(const f32x4*)(p + 16);
  return u.v;
}

static __device__ __forceinline__ v8f wmma_bf16(v16bf a, v16bf b, v8f c) {
  return __builtin_amdgcn_wmma_f32_16x16x32_bf16(false, a, false, b, (short)0, c,
                                                 false, false);
}

// ---------------------------------------------------------------- convert ----
__global__ void k_convert(const float* __restrict__ x,  const float* __restrict__ Wq,
                          const float* __restrict__ Wk, const float* __restrict__ Wv,
                          bf16_t* __restrict__ Xbf, bf16_t* __restrict__ Wbf) {
  const int t   = blockIdx.x * blockDim.x + threadIdx.x;
  const int nth = gridDim.x * blockDim.x;
  const int NX = NTOK * D_MODEL;
  for (int i = t; i < NX; i += nth) Xbf[i] = f2bf(x[i]);
  const int NW = D_MODEL * D_MODEL;
  for (int i = t; i < NW; i += nth) {
    Wbf[i]          = f2bf(Wq[i]);
    Wbf[NW + i]     = f2bf(Wk[i]);
    Wbf[2 * NW + i] = f2bf(Wv[i]);
  }
}

// --------------------------------------------------------------- QKV GEMM ----
// out[g][n][m] = sum_d X[n][d] * W_g[m][d] + b_g[m], written head-split as
// [g][b][h][s][Dh] in bf16.  128x128 block tile, 8 waves, 2x4 WMMA per wave.
// Double-buffered LDS with async global->LDS staging for tile i+1 overlapped
// with WMMA on tile i.
__global__ __launch_bounds__(256)
void k_qkv_gemm(const bf16_t* __restrict__ Xbf, const bf16_t* __restrict__ Wbf,
                const float* __restrict__ bq, const float* __restrict__ bk,
                const float* __restrict__ bv, bf16_t* __restrict__ QKVh) {
  __shared__ bf16_t Xs[2][128][40];
  __shared__ bf16_t Ws[2][128][40];
  const int g = blockIdx.z;
  const bf16_t* W    = Wbf + (size_t)g * D_MODEL * D_MODEL;
  const float*  bias = (g == 0) ? bq : (g == 1) ? bk : bv;
  bf16_t* outp = QKVh + (size_t)g * NTOK * D_MODEL;
  const int n0 = blockIdx.x * 128;   // output feature base
  const int m0 = blockIdx.y * 128;   // token base
  const int tid = threadIdx.x;
  const int lane = tid & 31, wave = tid >> 5;
  const int wm = wave & 3, wn = wave >> 2;   // wave -> (32-row, 64-col) subtile

  v8f acc[2][4];
#pragma unroll
  for (int i = 0; i < 2; ++i)
#pragma unroll
    for (int j = 0; j < 4; ++j) acc[i][j] = vzero8();

  const int row = tid >> 1, half = tid & 1;
  const bf16_t* gxr = Xbf + (size_t)(m0 + row) * D_MODEL + half * 16;
  const bf16_t* gwr = W   + (size_t)(n0 + row) * D_MODEL + half * 16;

  // prologue: stage tile 0 into buffer 0
  cp16_g2l(gxr,     &Xs[0][row][half * 16]);
  cp16_g2l(gxr + 8, &Xs[0][row][half * 16 + 8]);
  cp16_g2l(gwr,     &Ws[0][row][half * 16]);
  cp16_g2l(gwr + 8, &Ws[0][row][half * 16 + 8]);
  async_wait0();
  __syncthreads();

  const int nk = D_MODEL / 32;
  for (int it = 0; it < nk; ++it) {
    const int cur = it & 1;
    if (it + 1 < nk) {               // stage next tile into the other buffer
      const bf16_t* gx = gxr + (it + 1) * 32;
      const bf16_t* gw = gwr + (it + 1) * 32;
      cp16_g2l(gx,     &Xs[cur ^ 1][row][half * 16]);
      cp16_g2l(gx + 8, &Xs[cur ^ 1][row][half * 16 + 8]);
      cp16_g2l(gw,     &Ws[cur ^ 1][row][half * 16]);
      cp16_g2l(gw + 8, &Ws[cur ^ 1][row][half * 16 + 8]);
#if !HAVE_ASYNC_LDS
      __builtin_prefetch(gx + 32, 0, 1);
      __builtin_prefetch(gw + 32, 0, 1);
#endif
    }
    v16bf af[2], bfg[4];
#pragma unroll
    for (int mi = 0; mi < 2; ++mi)
      af[mi]  = load_frag(&Xs[cur][wm * 32 + mi * 16][0], 40, lane);
#pragma unroll
    for (int ni = 0; ni < 4; ++ni)
      bfg[ni] = load_frag(&Ws[cur][wn * 64 + ni * 16][0], 40, lane);
#pragma unroll
    for (int mi = 0; mi < 2; ++mi)
#pragma unroll
      for (int ni = 0; ni < 4; ++ni)
        acc[mi][ni] = wmma_bf16(af[mi], bfg[ni], acc[mi][ni]);
    async_wait0();
    __syncthreads();
  }

  const int colL = lane & 15, rbase = (lane >> 4) * 8;
#pragma unroll
  for (int mi = 0; mi < 2; ++mi) {
#pragma unroll
    for (int ni = 0; ni < 4; ++ni) {
      const int col = n0 + wn * 64 + ni * 16 + colL;
      const float bb = bias[col];
      const int h = col / HDIM, d = col % HDIM;
#pragma unroll
      for (int j = 0; j < 8; ++j) {
        const int rowg = m0 + wm * 32 + mi * 16 + rbase + j;
        const int b = rowg >> 11, s = rowg & (SEQ - 1);
        outp[((size_t)(b * NHEADS + h) * SEQ + s) * HDIM + d] =
            f2bf(acc[mi][ni][j] + bb);
      }
    }
  }
}

// -------------------------------------------- pass 1: per-column softmax stats
// For each key column k: M[k] = max_q w[q,k], R[k] = 1/sum_q exp(w[q,k]-M[k]).
// Each wave owns a 16-column strip; loops over q-tiles (q >= k region only).
__global__ __launch_bounds__(256)
void k_colstats(const bf16_t* __restrict__ QKVh, const unsigned char* __restrict__ pad,
                float* __restrict__ Mcol, float* __restrict__ Rcol) {
  const int bh = blockIdx.y;
  const int b = bh >> 2;
  const bf16_t* Q = QKVh + (size_t)bh * SEQ * HDIM;
  const bf16_t* K = QKVh + (size_t)NTOK * D_MODEL + (size_t)bh * SEQ * HDIM;
  const unsigned char* pm = pad + (size_t)b * SEQ;
  const int lane = threadIdx.x & 31, wave = threadIdx.x >> 5;
  const int c0 = blockIdx.x * 128 + wave * 16;

  v16bf kf[6];
#pragma unroll
  for (int c = 0; c < 6; ++c)
    kf[c] = load_frag(K + (size_t)c0 * HDIM + c * 32, HDIM, lane);

  float m_run = -3.0e38f, z_run = 0.0f;
  const int kcol  = c0 + (lane & 15);
  const int rbase = (lane >> 4) * 8;
  for (int q0 = c0; q0 < SEQ; q0 += 16) {
    v8f sacc = vzero8();
#pragma unroll
    for (int c = 0; c < 6; ++c) {
      v16bf qf = load_frag(Q + (size_t)q0 * HDIM + c * 32, HDIM, lane);
      sacc = wmma_bf16(qf, kf[c], sacc);
    }
    float w[8]; float tmax = -3.0e38f;
#pragma unroll
    for (int j = 0; j < 8; ++j) {
      const int q = q0 + rbase + j;
      float v = sacc[j] * INV_SCALE;
      if ((kcol > q) || pm[q]) v = NEGV;
      w[j] = v; tmax = fmaxf(tmax, v);
    }
    tmax = fmaxf(tmax, __shfl_xor(tmax, 16));    // partner lane holds other 8 rows
    const float nm = fmaxf(m_run, tmax);
    float zs = 0.f;
#pragma unroll
    for (int j = 0; j < 8; ++j) zs += __expf(w[j] - nm);
    zs += __shfl_xor(zs, 16);
    z_run = z_run * __expf(m_run - nm) + zs;
    m_run = nm;
  }
  if (lane < 16) {
    Mcol[(size_t)bh * SEQ + kcol] = m_run;
    Rcol[(size_t)bh * SEQ + kcol] = 1.0f / z_run;
  }
}

// ------------------------------------------------- pass 2: P = softmax, P @ V
// Per block: 128 q rows (one 16-row tile per wave), shared K chunk staged
// async into LDS (all waves consume the same K tile), V staged transposed.
__global__ __launch_bounds__(256)
void k_attn_pv(const bf16_t* __restrict__ QKVh, const unsigned char* __restrict__ pad,
               const float* __restrict__ Mcol, const float* __restrict__ Rcol,
               float* __restrict__ attnout) {
  __shared__ bf16_t Ks[32][200];      // K chunk row-major: Ks[k][d]
  __shared__ bf16_t Vt[HDIM][32];     // V chunk transposed: Vt[d][k]
  __shared__ bf16_t Pl[8][16][40];    // per-wave P tile (C-layout -> A-layout)
  const int bh = blockIdx.y;
  const int b = bh >> 2, h = bh & 3;
  const bf16_t* Q = QKVh + (size_t)bh * SEQ * HDIM;
  const bf16_t* K = QKVh + (size_t)NTOK * D_MODEL + (size_t)bh * SEQ * HDIM;
  const bf16_t* V = QKVh + 2 * (size_t)NTOK * D_MODEL + (size_t)bh * SEQ * HDIM;
  const float* Mc = Mcol + (size_t)bh * SEQ;
  const float* Rc = Rcol + (size_t)bh * SEQ;
  const unsigned char* pm = pad + (size_t)b * SEQ;
  const int lane = threadIdx.x & 31, wave = threadIdx.x >> 5;
  const int q0 = blockIdx.x * 128 + wave * 16;

  v16bf qf[6];
#pragma unroll
  for (int c = 0; c < 6; ++c)
    qf[c] = load_frag(Q + (size_t)q0 * HDIM + c * 32, HDIM, lane);

  v8f acc[12];
#pragma unroll
  for (int i = 0; i < 12; ++i) acc[i] = vzero8();

  const int kmax  = blockIdx.x * 128 + 128;      // k <= q within this block
  const int colL  = lane & 15, rbase = (lane >> 4) * 8;
  const int kr  = threadIdx.x >> 3;              // staging: k row 0..31
  const int seg = (threadIdx.x & 7) * 24;        // staging: d base (24 elems)

  for (int k0 = 0; k0 < kmax; k0 += 32) {
    __syncthreads();                             // prior reads of Ks/Vt done
    {
      const bf16_t* ksrc = K + (size_t)(k0 + kr) * HDIM + seg;
      cp16_g2l(ksrc,      &Ks[kr][seg]);
      cp16_g2l(ksrc + 8,  &Ks[kr][seg + 8]);
      cp16_g2l(ksrc + 16, &Ks[kr][seg + 16]);
      const bf16_t* vsrc = V + (size_t)(k0 + kr) * HDIM + seg;
#pragma unroll
      for (int e = 0; e < 24; ++e) Vt[seg + e][kr] = vsrc[e];
    }
    async_wait0();
    __syncthreads();
    if (k0 <= q0 + 15) {                         // wave-uniform: EXEC stays full
      v8f s0 = vzero8(), s1 = vzero8();
#pragma unroll
      for (int c = 0; c < 6; ++c) {
        v16bf kf0 = load_frag(&Ks[0][c * 32], 200, lane);
        v16bf kf1 = load_frag(&Ks[16][c * 32], 200, lane);
        s0 = wmma_bf16(qf[c], kf0, s0);
        s1 = wmma_bf16(qf[c], kf1, s1);
      }
#pragma unroll
      for (int t = 0; t < 2; ++t) {
        const int kc = k0 + t * 16 + colL;
        const float M = Mc[kc], R = Rc[kc];
#pragma unroll
        for (int j = 0; j < 8; ++j) {
          const int q = q0 + rbase + j;
          float w = (t ? s1[j] : s0[j]) * INV_SCALE;
          if ((kc > q) || pm[q]) w = NEGV;
          Pl[wave][rbase + j][t * 16 + colL] = f2bf(__expf(w - M) * R);
        }
      }
      // wave-local LDS fence: P stores visible before cross-lane frag reads
#if __has_builtin(__builtin_amdgcn_s_wait_dscnt)
      __builtin_amdgcn_s_wait_dscnt(0);
#else
      asm volatile("s_wait_dscnt 0" ::: "memory");
#endif
      v16bf pf = load_frag(&Pl[wave][0][0], 40, lane);
#pragma unroll
      for (int ni = 0; ni < 12; ++ni) {
        v16bf vf = load_frag(&Vt[ni * 16][0], 32, lane);
        acc[ni] = wmma_bf16(pf, vf, acc[ni]);
      }
    }
  }

#pragma unroll
  for (int ni = 0; ni < 12; ++ni) {
    const int d = ni * 16 + colL;
#pragma unroll
    for (int j = 0; j < 8; ++j) {
      const int q = q0 + rbase + j;
      attnout[((size_t)(b * SEQ + q)) * D_MODEL + h * HDIM + d] = acc[ni][j];
    }
  }
}

// ------------------------------------------------------ residual + LayerNorm -
static __device__ __forceinline__ float block_sum(float v, float* red) {
#pragma unroll
  for (int o = 16; o > 0; o >>= 1) v += __shfl_xor(v, o);
  const int lane = threadIdx.x & 31, wave = threadIdx.x >> 5;
  if (lane == 0) red[wave] = v;
  __syncthreads();
  float r = (threadIdx.x < 8) ? red[threadIdx.x] : 0.0f;
  if (wave == 0) {
#pragma unroll
    for (int o = 4; o > 0; o >>= 1) r += __shfl_xor(r, o);
    if (lane == 0) red[0] = r;
  }
  __syncthreads();
  const float tot = red[0];
  __syncthreads();
  return tot;
}

__global__ __launch_bounds__(256)
void k_ln(const float* __restrict__ x, const float* __restrict__ attn,
          const float* __restrict__ gamma, const float* __restrict__ beta,
          float* __restrict__ out) {
  __shared__ float red[8];
  const int row = blockIdx.x;
  const float* xr = x + (size_t)row * D_MODEL;
  const float* ar = attn + (size_t)row * D_MODEL;
  const int t = threadIdx.x;
  float y[3]; float s = 0.f;
#pragma unroll
  for (int e = 0; e < 3; ++e) { const int c = t + e * 256; y[e] = xr[c] + ar[c]; s += y[e]; }
  const float mu = block_sum(s, red) * (1.0f / D_MODEL);
  float vs = 0.f;
#pragma unroll
  for (int e = 0; e < 3; ++e) { const float d = y[e] - mu; vs += d * d; }
  const float rstd = rsqrtf(block_sum(vs, red) * (1.0f / D_MODEL) + LN_EPS);
#pragma unroll
  for (int e = 0; e < 3; ++e) {
    const int c = t + e * 256;
    out[(size_t)row * D_MODEL + c] = (y[e] - mu) * rstd * gamma[c] + beta[c];
  }
}

// ------------------------------------------------------------------- launch --
extern "C" void kernel_launch(void* const* d_in, const int* in_sizes, int n_in,
                              void* d_out, int out_size, void* d_ws, size_t ws_size,
                              hipStream_t stream) {
  const float* x  = (const float*)d_in[0];
  const unsigned char* amask = (const unsigned char*)d_in[1];   // bool (B,1,S)
  const float* Wq = (const float*)d_in[2];
  const float* bq = (const float*)d_in[3];
  const float* Wk = (const float*)d_in[4];
  const float* bk = (const float*)d_in[5];
  const float* Wv = (const float*)d_in[6];
  const float* bv = (const float*)d_in[7];
  const float* gamma = (const float*)d_in[8];
  const float* beta  = (const float*)d_in[9];
  float* out = (float*)d_out;

  char* ws = (char*)d_ws;
  size_t o = 0;
  bf16_t* Xbf  = (bf16_t*)(ws + o); o += (size_t)NTOK * D_MODEL * 2;        // 12.6 MB
  bf16_t* Wbf  = (bf16_t*)(ws + o); o += (size_t)3 * D_MODEL * D_MODEL * 2; //  3.5 MB
  bf16_t* QKVh = (bf16_t*)(ws + o); o += (size_t)3 * NTOK * D_MODEL * 2;    // 37.7 MB
  float*  Mcol = (float*)(ws + o);  o += (size_t)BATCH * NHEADS * SEQ * 4;
  float*  Rcol = (float*)(ws + o);  o += (size_t)BATCH * NHEADS * SEQ * 4;
  float*  attnout = (float*)(ws + o); o += (size_t)NTOK * D_MODEL * 4;      // 25.2 MB

  k_convert<<<1024, 256, 0, stream>>>(x, Wq, Wk, Wv, Xbf, Wbf);
  k_qkv_gemm<<<dim3(D_MODEL / 128, NTOK / 128, 3), 256, 0, stream>>>(
      Xbf, Wbf, bq, bk, bv, QKVh);
  k_colstats<<<dim3(SEQ / 128, BATCH * NHEADS), 256, 0, stream>>>(
      QKVh, amask, Mcol, Rcol);
  k_attn_pv<<<dim3(SEQ / 128, BATCH * NHEADS), 256, 0, stream>>>(
      QKVh, amask, Mcol, Rcol, attnout);
  k_ln<<<NTOK, 256, 0, stream>>>(x, attnout, gamma, beta, out);
}
